// MaskedMultiheadAttention_51591147160164
// MI455X (gfx1250) — matrix operand order repaired and checked by reading, not compiled, not run on hardware
//
#include <hip/hip_runtime.h>
#include <hip/hip_bf16.h>

// Problem constants (match reference)
#define BB   4
#define SS   1000
#define DD   1024
#define NHH  16
#define DKK  64
#define MM   (BB * SS)          // 4000 rows in flattened [B*S, D]
#define XE   ((size_t)MM * DD)  // 4,096,000 elements per activation tensor
#define WE   ((size_t)DD * DD)  // 1,048,576 elements per weight matrix

typedef __attribute__((ext_vector_type(16))) __bf16 v16bf;
typedef __attribute__((ext_vector_type(8)))  float  v8f;

union FragU {
  v16bf           f;
  uint4           q[2];
  unsigned short  u[16];
};

static __device__ __forceinline__ unsigned short f2bf(float x) {
  unsigned u = __float_as_uint(x);
  u += 0x7FFFu + ((u >> 16) & 1u);   // round-to-nearest-even
  return (unsigned short)(u >> 16);
}

#define NEG_INF (-__builtin_inff())

// ---------------------------------------------------------------------------
// f32 -> bf16 elementwise convert
// ---------------------------------------------------------------------------
__global__ void __launch_bounds__(256)
cvt_f32_bf16(const float* __restrict__ src, unsigned short* __restrict__ dst, int n) {
  int i = blockIdx.x * blockDim.x + threadIdx.x;
  int stride = gridDim.x * blockDim.x;
  for (; i < n; i += stride) dst[i] = f2bf(src[i]);
}

// ---------------------------------------------------------------------------
// Y = X @ W^T + bias.   X: bf16 [M x K] row-major, W: bf16 [N x K] row-major.
// mode 0: store bf16 row-major [M x N]
// mode 1: store bf16 transposed per-head: Vt[b][h][d][s]  (n -> h,d ; m -> b,s)
// mode 2: store f32 row-major [M x N]
// grid.x = M/16 blocks, 256 threads (8 waves); wave w owns the 16x128 strip
// at n0 = 128*w.  K loop in steps of 32 -> v_wmma_f32_16x16x32_bf16.
// ---------------------------------------------------------------------------
__global__ void __launch_bounds__(256)
gemm_xwt(const unsigned short* __restrict__ Ab,
         const unsigned short* __restrict__ Wb,
         const float* __restrict__ bias,
         void* __restrict__ Out, int M, int N, int K, int mode)
{
  const int lane = threadIdx.x & 31;
  const int wave = threadIdx.x >> 5;
  const int m0   = blockIdx.x * 16;
  const int n0   = wave * 128;
  const int col  = lane & 15;
  const int hi   = lane >> 4;

  v8f c[8] = {};

  const unsigned short* arow = Ab + (size_t)(m0 + col) * K + hi * 8;

  for (int k = 0; k < K; k += 32) {
    FragU a;
    a.q[0] = *(const uint4*)(arow + k);        // K = k+hi*8   .. +7
    a.q[1] = *(const uint4*)(arow + k + 16);   // K = k+16+hi*8.. +7
#pragma unroll
    for (int s = 0; s < 8; ++s) {
      const unsigned short* bp = Wb + (size_t)(n0 + s * 16 + col) * K + k + hi * 8;
      FragU b;
      b.q[0] = *(const uint4*)bp;              // K = k+hi*8   .. +7
      b.q[1] = *(const uint4*)(bp + 16);       // K = k+16+hi*8.. +7
      c[s] = __builtin_amdgcn_wmma_f32_16x16x32_bf16(
                 false, a.f, false, b.f, (short)0, c[s], false, false);
    }
  }

  if (mode == 0) {
    unsigned short* Y = (unsigned short*)Out;
#pragma unroll
    for (int s = 0; s < 8; ++s) {
      int n = n0 + s * 16 + col;
      float bv = bias[n];
#pragma unroll
      for (int r = 0; r < 8; ++r) {
        int m = m0 + r + 8 * hi;
        Y[(size_t)m * N + n] = f2bf(c[s][r] + bv);
      }
    }
  } else if (mode == 1) {
    unsigned short* Vt = (unsigned short*)Out;
#pragma unroll
    for (int s = 0; s < 8; ++s) {
      int n = n0 + s * 16 + col;
      int h = n >> 6, d = n & 63;
      float bv = bias[n];
#pragma unroll
      for (int r = 0; r < 8; ++r) {
        int m = m0 + r + 8 * hi;
        int b = m / SS, ss = m - b * SS;
        Vt[((size_t)(b * NHH + h) * DKK + d) * SS + ss] = f2bf(c[s][r] + bv);
      }
    }
  } else {
    float* Y = (float*)Out;
#pragma unroll
    for (int s = 0; s < 8; ++s) {
      int n = n0 + s * 16 + col;
      float bv = bias[n];
#pragma unroll
      for (int r = 0; r < 8; ++r) {
        int m = m0 + r + 8 * hi;
        Y[(size_t)m * N + n] = c[s][r] + bv;
      }
    }
  }
}

// ---------------------------------------------------------------------------
// Raw masked scores: A[b][h][q][kk] = 0.125 * Q.K  (or -inf if masked).
// grid.x = B*NH*63 ; block 256 = 8 waves (wave = 128-wide kk strip).
// ---------------------------------------------------------------------------
__global__ void __launch_bounds__(256)
attn_scores(const unsigned short* __restrict__ Qb,
            const unsigned short* __restrict__ Kb,
            const int* __restrict__ mask,
            float* __restrict__ A)
{
  const int mt   = blockIdx.x % 63;
  const int bh   = blockIdx.x / 63;
  const int b    = bh >> 4, h = bh & 15;
  const int lane = threadIdx.x & 31;
  const int wave = threadIdx.x >> 5;
  const int n0   = wave * 128;
  const int col  = lane & 15, hi = lane >> 4;
  const int q0   = mt * 16;

  int qa = q0 + col; if (qa > SS - 1) qa = SS - 1;   // clamp loads, mask at store
  const unsigned short* arow = Qb + (size_t)(b * SS + qa) * DD + h * DKK + hi * 8;

  v8f c[8] = {};

#pragma unroll
  for (int k = 0; k < DKK; k += 32) {
    FragU a;
    a.q[0] = *(const uint4*)(arow + k);
    a.q[1] = *(const uint4*)(arow + k + 16);
#pragma unroll
    for (int s = 0; s < 8; ++s) {
      int kk = n0 + s * 16 + col; if (kk > SS - 1) kk = SS - 1;
      const unsigned short* bp = Kb + (size_t)(b * SS + kk) * DD + h * DKK + k + hi * 8;
      FragU bb;
      bb.q[0] = *(const uint4*)bp;
      bb.q[1] = *(const uint4*)(bp + 16);
      c[s] = __builtin_amdgcn_wmma_f32_16x16x32_bf16(
                 false, a.f, false, bb.f, (short)0, c[s], false, false);
    }
  }

  const float scale = 0.125f;  // 1/sqrt(DK)
  float* Ab = A + (size_t)(b * NHH + h) * SS * SS;
#pragma unroll
  for (int s = 0; s < 8; ++s) {
    int kk = n0 + s * 16 + col;
    if (kk >= SS) continue;
    int mk = mask[b * SS + kk];
#pragma unroll
    for (int r = 0; r < 8; ++r) {
      int q = q0 + r + 8 * hi;
      if (q >= SS) continue;
      float v = c[s][r] * scale;
      if (kk > q || mk == 0) v = NEG_INF;
      Ab[(size_t)q * SS + kk] = v;
    }
  }
}

// ---------------------------------------------------------------------------
// In-place row softmax over rows of length SS.  grid.x = B*NH*S, block 256.
// ---------------------------------------------------------------------------
__global__ void __launch_bounds__(256)
softmax_rows(float* __restrict__ A)
{
  const int tid = threadIdx.x;
  float* row = A + (size_t)blockIdx.x * SS;
  float v[4];
  float m = NEG_INF;
#pragma unroll
  for (int j = 0; j < 4; ++j) {
    int i = tid + j * 256;
    v[j] = (i < SS) ? row[i] : NEG_INF;
    m = fmaxf(m, v[j]);
  }
  __shared__ float red[256];
  red[tid] = m; __syncthreads();
  for (int off = 128; off > 0; off >>= 1) {
    if (tid < off) red[tid] = fmaxf(red[tid], red[tid + off]);
    __syncthreads();
  }
  float rm = red[0];
  __syncthreads();
  float s = 0.f;
#pragma unroll
  for (int j = 0; j < 4; ++j) {
    int i = tid + j * 256;
    float e = (i < SS) ? __expf(v[j] - rm) : 0.f;
    v[j] = e; s += e;
  }
  red[tid] = s; __syncthreads();
  for (int off = 128; off > 0; off >>= 1) {
    if (tid < off) red[tid] += red[tid + off];
    __syncthreads();
  }
  float inv = 1.f / red[0];
#pragma unroll
  for (int j = 0; j < 4; ++j) {
    int i = tid + j * 256;
    if (i < SS) row[i] = v[j] * inv;
  }
}

// ---------------------------------------------------------------------------
// Context: Ctx[b][s][h*64+d] = sum_kk A[b][h][q][kk] * Vt[b][h][d][kk].
// grid.x = B*NH*8, block 256 (8 waves); wave owns a 16(q) x 64(d) strip.
// A (f32) converted to bf16 on the fly; out-of-range kk zeros the A frag so
// neighboring-row garbage in B contributes nothing.
// ---------------------------------------------------------------------------
__global__ void __launch_bounds__(256)
attn_context(const float* __restrict__ A,
             const unsigned short* __restrict__ Vt,
             unsigned short* __restrict__ Ctx)
{
  const int blk  = blockIdx.x & 7;
  const int bh   = blockIdx.x >> 3;
  const int b    = bh >> 4, h = bh & 15;
  const int lane = threadIdx.x & 31;
  const int wave = threadIdx.x >> 5;
  const int mt   = blk * 8 + wave;      // 0..63 (63 used)
  const int q0   = mt * 16;
  const int col  = lane & 15, hi = lane >> 4;

  int qa = q0 + col; if (qa > SS - 1) qa = SS - 1;
  const float* Arow = A + ((size_t)(b * NHH + h) * SS + qa) * SS;
  const unsigned short* Vbh = Vt + (size_t)(b * NHH + h) * DKK * SS;

  v8f c[4] = {};

  for (int k = 0; k < SS; k += 32) {
    FragU a;
    if (k + 32 <= SS) {
      const float* p0 = Arow + k + hi * 8;
#pragma unroll
      for (int j = 0; j < 8; ++j) a.u[j]     = f2bf(p0[j]);
#pragma unroll
      for (int j = 0; j < 8; ++j) a.u[8 + j] = f2bf(p0[16 + j]);
    } else {
#pragma unroll
      for (int j = 0; j < 8; ++j) {
        int k0 = k + hi * 8 + j;
        int k1 = k0 + 16;
        a.u[j]     = (k0 < SS) ? f2bf(Arow[k0]) : (unsigned short)0;
        a.u[8 + j] = (k1 < SS) ? f2bf(Arow[k1]) : (unsigned short)0;
      }
    }
#pragma unroll
    for (int s = 0; s < 4; ++s) {
      int d = s * 16 + col;
      const unsigned short* bp = Vbh + (size_t)d * SS + k + hi * 8;
      FragU bb;
      bb.q[0] = *(const uint4*)bp;       // may overread row tail; A frag is 0 there
      bb.q[1] = *(const uint4*)(bp + 16);
      c[s] = __builtin_amdgcn_wmma_f32_16x16x32_bf16(
                 false, a.f, false, bb.f, (short)0, c[s], false, false);
    }
  }

#pragma unroll
  for (int s = 0; s < 4; ++s) {
    int d = s * 16 + col;
#pragma unroll
    for (int r = 0; r < 8; ++r) {
      int q = q0 + r + 8 * hi;
      if (q >= SS) continue;
      Ctx[(size_t)(b * SS + q) * DD + h * DKK + d] = f2bf(c[s][r]);
    }
  }
}

// ---------------------------------------------------------------------------
extern "C" void kernel_launch(void* const* d_in, const int* in_sizes, int n_in,
                              void* d_out, int out_size, void* d_ws, size_t ws_size,
                              hipStream_t stream) {
  (void)in_sizes; (void)n_in; (void)out_size; (void)ws_size;

  const float* q    = (const float*)d_in[0];
  const float* k    = (const float*)d_in[1];
  const float* v    = (const float*)d_in[2];
  const int*   mask = (const int*)  d_in[3];
  const float* Wq   = (const float*)d_in[4];
  const float* bq   = (const float*)d_in[5];
  const float* Wk   = (const float*)d_in[6];
  const float* bk   = (const float*)d_in[7];
  const float* Wv   = (const float*)d_in[8];
  const float* bv   = (const float*)d_in[9];
  const float* Wp   = (const float*)d_in[10];
  const float* bp   = (const float*)d_in[11];

  float* Xout = (float*)d_out;                 // [B,S,D]
  float* Aout = (float*)d_out + XE;            // [B,NH,S,S]

  // workspace carve-up (all bf16 buffers, 256B aligned)
  size_t off = 0;
  auto carve = [&](size_t bytes) {
    void* p = (char*)d_ws + off;
    off += (bytes + 255) & ~(size_t)255;
    return p;
  };
  unsigned short* qb  = (unsigned short*)carve(XE * 2);
  unsigned short* kb  = (unsigned short*)carve(XE * 2);
  unsigned short* vb  = (unsigned short*)carve(XE * 2);
  unsigned short* wqb = (unsigned short*)carve(WE * 2);
  unsigned short* wkb = (unsigned short*)carve(WE * 2);
  unsigned short* wvb = (unsigned short*)carve(WE * 2);
  unsigned short* wpb = (unsigned short*)carve(WE * 2);
  unsigned short* Qb  = (unsigned short*)carve(XE * 2);
  unsigned short* Kb  = (unsigned short*)carve(XE * 2);
  unsigned short* Vtb = (unsigned short*)carve(XE * 2);   // [B,NH,DK,S]
  unsigned short* Ctx = (unsigned short*)carve(XE * 2);   // overread pad for Vtb tail

  // 1) convert inputs + weights to bf16
  cvt_f32_bf16<<<2048, 256, 0, stream>>>(q,  qb,  (int)XE);
  cvt_f32_bf16<<<2048, 256, 0, stream>>>(k,  kb,  (int)XE);
  cvt_f32_bf16<<<2048, 256, 0, stream>>>(v,  vb,  (int)XE);
  cvt_f32_bf16<<<1024, 256, 0, stream>>>(Wq, wqb, (int)WE);
  cvt_f32_bf16<<<1024, 256, 0, stream>>>(Wk, wkb, (int)WE);
  cvt_f32_bf16<<<1024, 256, 0, stream>>>(Wv, wvb, (int)WE);
  cvt_f32_bf16<<<1024, 256, 0, stream>>>(Wp, wpb, (int)WE);

  // 2) projections
  gemm_xwt<<<MM / 16, 256, 0, stream>>>(qb, wqb, bq, Qb,  MM, DD, DD, 0);
  gemm_xwt<<<MM / 16, 256, 0, stream>>>(kb, wkb, bk, Kb,  MM, DD, DD, 0);
  gemm_xwt<<<MM / 16, 256, 0, stream>>>(vb, wvb, bv, Vtb, MM, DD, DD, 1);

  // 3) masked raw scores -> A region of d_out
  attn_scores<<<BB * NHH * 63, 256, 0, stream>>>(Qb, Kb, mask, Aout);

  // 4) in-place row softmax
  softmax_rows<<<BB * NHH * SS, 256, 0, stream>>>(Aout);

  // 5) context = A @ V
  attn_context<<<BB * NHH * 8, 256, 0, stream>>>(Aout, Vtb, Ctx);

  // 6) output projection (f32 store)
  gemm_xwt<<<MM / 16, 256, 0, stream>>>(Ctx, wpb, bp, Xout, MM, DD, DD, 2);
}
